// AttentionLayer_13134009991917
// MI455X (gfx1250) — matrix operand order repaired and checked by reading, not compiled, not run on hardware
//
#include <hip/hip_runtime.h>
#include <hip/hip_bf16.h>

typedef __bf16 bf16_t;
typedef bf16_t v16bf __attribute__((ext_vector_type(16)));
typedef float  v8f   __attribute__((ext_vector_type(8)));

union V16 {
    uint4  q[2];
    bf16_t h[16];
    v16bf  v;
};

// ---------------------------------------------------------------------------
// WMMA fragment builders for V_WMMA_F32_16X16X32_BF16 (wave32).
// A (16x32, MxK) per-lane: m = lane&15 ; K = g + 0..15,
//   g = (lane<16) ? 0 : 16.
// B (32x16, KxN) per-lane: n = lane&15 ; K = kb + 0..15, kb = (lane<16)?0:16.
// D (16x16 f32) per-lane: VGPR r -> m = r + ((lane<16)?0:8), n = lane&15.
// ---------------------------------------------------------------------------

// A fragment from row-major f32 matrix W (ld = row stride in elems).
__device__ __forceinline__ v16bf a_frag_f32(const float* __restrict__ W,
                                            int row_base, int ld, int kk, int lane) {
    const int row = row_base + (lane & 15);
    const int g   = (lane & 16) ? 16 : 0;
    const float* p = W + (size_t)row * ld + kk + g;
    V16 u;
#pragma unroll
    for (int h = 0; h < 8; ++h) u.h[h]     = (bf16_t)p[h];
#pragma unroll
    for (int h = 0; h < 8; ++h) u.h[8 + h] = (bf16_t)p[8 + h];
    return u.v;
}

// B fragment from f32 matrix X[k][n] with ld between K rows (strided gather).
__device__ __forceinline__ v16bf b_frag_f32(const float* __restrict__ X,
                                            int ld, int kk, int n_base, int lane) {
    const int n  = n_base + (lane & 15);
    const int kb = (lane & 16) ? 16 : 0;
    const float* p = X + (size_t)(kk + kb) * ld + n;
    V16 u;
#pragma unroll
    for (int h = 0; h < 16; ++h) u.h[h] = (bf16_t)p[(size_t)h * ld];
    return u.v;
}

// A fragment from bf16 matrix, K contiguous within a row (ld = row stride).
__device__ __forceinline__ v16bf a_frag_bf16(const bf16_t* base,
                                             int row_base, int ld, int kk, int lane) {
    const int row = row_base + (lane & 15);
    const int g   = (lane & 16) ? 16 : 0;
    const bf16_t* p = base + (size_t)row * ld + kk + g;
    V16 u;
    u.q[0] = *reinterpret_cast<const uint4*>(p);        // K-run 0 (8 x bf16)
    u.q[1] = *reinterpret_cast<const uint4*>(p + 8);    // K-run 1
    return u.v;
}

// B fragment from bf16 matrix stored [n][k] with K contiguous (ld = row stride).
__device__ __forceinline__ v16bf b_frag_bf16(const bf16_t* base,
                                             int n_base, int ld, int kk, int lane) {
    const int n  = n_base + (lane & 15);
    const int kb = (lane & 16) ? 16 : 0;
    const bf16_t* p = base + (size_t)n * ld + kk + kb;
    V16 u;
    u.q[0] = *reinterpret_cast<const uint4*>(p);
    u.q[1] = *reinterpret_cast<const uint4*>(p + 8);
    return u.v;
}

__device__ __forceinline__ v8f wmma_bf16(v16bf a, v16bf b, v8f c) {
    return __builtin_amdgcn_wmma_f32_16x16x32_bf16(false, a, false, b, (short)0, c, false, false);
}

// ---------------------------------------------------------------------------
// Kernel 1/2: Y_bf16[b][o][t] = tanh( sum_f W[o][f] * X[b][f][t] + bias[o] )
// O = 128, F template (128). grid = (T/16, B), block = 256 (8 waves).
// Wave w computes the 16-row o-strip [16w, 16w+16) x 16 t columns.
// ---------------------------------------------------------------------------
template <int F>
__global__ void __launch_bounds__(256)
dense_tanh_bf16_kernel(const float* __restrict__ W, const float* __restrict__ bias,
                       const float* __restrict__ X, bf16_t* __restrict__ Y, int T) {
    const int lane = threadIdx.x & 31;
    const int w    = threadIdx.x >> 5;
    const int t0   = blockIdx.x * 16;
    const int b    = blockIdx.y;
    const float* Xb = X + (size_t)b * F * T;

    v8f acc = {0.f, 0.f, 0.f, 0.f, 0.f, 0.f, 0.f, 0.f};
#pragma unroll
    for (int kk = 0; kk < F; kk += 32) {
        v16bf a  = a_frag_f32(W, 16 * w, F, kk, lane);
        v16bf bb = b_frag_f32(Xb, T, kk, t0, lane);
        acc = wmma_bf16(a, bb, acc);
    }
    const int hi = (lane & 16) ? 8 : 0;
    const int t  = t0 + (lane & 15);
#pragma unroll
    for (int r = 0; r < 8; ++r) {
        const int o = 16 * w + r + hi;
        Y[((size_t)b * 128 + o) * T + t] = (bf16_t)tanhf(acc[r] + bias[o]);
    }
}

// ---------------------------------------------------------------------------
// Kernel 3: S_part[ks][b][d][m] = sum_{t in chunk ks} k[b][d][t] * v[b][m][t]
// grid = (8 d-strips, B, 8 k-splits), block = 256; wave w -> m-tile w.
// ---------------------------------------------------------------------------
__global__ void __launch_bounds__(256)
s_partial_kernel(const bf16_t* __restrict__ Kmat, const bf16_t* __restrict__ Vmat,
                 float* __restrict__ S_part, int T, int TCH) {
    const int lane   = threadIdx.x & 31;
    const int w      = threadIdx.x >> 5;   // m-tile
    const int dstrip = blockIdx.x;         // 0..7
    const int b      = blockIdx.y;
    const int ks     = blockIdx.z;         // 0..7
    const bf16_t* Kb = Kmat + (size_t)b * 128 * T;
    const bf16_t* Vb = Vmat + (size_t)b * 128 * T;
    const int tbase  = ks * TCH;

    v8f acc = {0.f, 0.f, 0.f, 0.f, 0.f, 0.f, 0.f, 0.f};
    for (int kk = 0; kk < TCH; kk += 32) {
        v16bf a  = a_frag_bf16(Kb, dstrip * 16, T, tbase + kk, lane);  // A[m=d][k=t]
        v16bf bb = b_frag_bf16(Vb, w * 16,      T, tbase + kk, lane);  // B[k=t][n=m]
        acc = wmma_bf16(a, bb, acc);
    }
    const int hi = (lane & 16) ? 8 : 0;
    const int m  = 16 * w + (lane & 15);
    float* outp = S_part + ((size_t)ks * 8 + b) * 128 * 128;
#pragma unroll
    for (int r = 0; r < 8; ++r) {
        const int d = dstrip * 16 + r + hi;
        outp[(size_t)d * 128 + m] = acc[r];
    }
}

// Kernel 4: S[i] = sum_s S_part[s][i], i over B*128*128.
__global__ void __launch_bounds__(256)
s_reduce_kernel(const float* __restrict__ S_part, float* __restrict__ S, int n) {
    const int i = blockIdx.x * blockDim.x + threadIdx.x;
    if (i < n) {
        float s = 0.f;
#pragma unroll
        for (int ks = 0; ks < 8; ++ks) s += S_part[(size_t)ks * n + i];
        S[i] = s;
    }
}

// Kernel 5: W2[b][m][d] = sum_{m'} Wa[m][m'] * S[b][d][m']   (tiny, exact f32)
__global__ void __launch_bounds__(256)
w2_kernel(const float* __restrict__ Wa, const float* __restrict__ S,
          float* __restrict__ W2) {
    const int b   = blockIdx.y;
    const int idx = blockIdx.x * 256 + threadIdx.x;   // 0..16383
    const int m = idx >> 7, d = idx & 127;
    const float* Sb = S + (size_t)b * 128 * 128;      // S[d][m']
    float acc = 0.f;
    for (int mp = 0; mp < 128; ++mp)
        acc += Wa[m * 128 + mp] * Sb[(size_t)d * 128 + mp];
    W2[((size_t)b * 128 + m) * 128 + d] = acc;
}

// ---------------------------------------------------------------------------
// Kernel 6 (fused): q_tile = tanh(Wq * query_tile + bq)  [128 x 16, -> LDS bf16]
//                   out    = tanh(W2[b] * q_tile + ba)   [128 x 16]
// grid = (T/16, B), block = 256 (8 waves). q never touches HBM.
// ---------------------------------------------------------------------------
__global__ void __launch_bounds__(256)
attn_out_kernel(const float* __restrict__ Wq, const float* __restrict__ bq,
                const float* __restrict__ query, const float* __restrict__ W2,
                const float* __restrict__ ba, float* __restrict__ out, int T) {
    __shared__ __align__(16) bf16_t qt[16][128];   // [t][d], 4 KB

    const int lane = threadIdx.x & 31;
    const int w    = threadIdx.x >> 5;
    const int t0   = blockIdx.x * 16;
    const int b    = blockIdx.y;
    const int hi   = (lane & 16) ? 8 : 0;
    const int tl   = lane & 15;

    // ---- stage 1: q rows [16w, 16w+16) for 16 t columns ----
    const float* Qb = query + (size_t)b * 256 * T;
    v8f acc = {0.f, 0.f, 0.f, 0.f, 0.f, 0.f, 0.f, 0.f};
#pragma unroll
    for (int kk = 0; kk < 256; kk += 32) {
        v16bf a  = a_frag_f32(Wq, 16 * w, 256, kk, lane);
        v16bf bb = b_frag_f32(Qb, T, kk, t0, lane);
        acc = wmma_bf16(a, bb, acc);
    }
    // lane holds d = 16w + hi + r, r = 0..7 (contiguous) -> one b128 LDS store
    __align__(16) bf16_t tmp[8];
#pragma unroll
    for (int r = 0; r < 8; ++r) {
        const int d = 16 * w + hi + r;
        tmp[r] = (bf16_t)tanhf(acc[r] + bq[d]);
    }
    *reinterpret_cast<uint4*>(&qt[tl][16 * w + hi]) = *reinterpret_cast<const uint4*>(tmp);
    __syncthreads();

    // ---- stage 2: out rows [16w, 16w+16) = tanh(W2 * q_tile + ba) ----
    const float* W2b = W2 + (size_t)b * 128 * 128;   // [m][d]
    v8f acc2 = {0.f, 0.f, 0.f, 0.f, 0.f, 0.f, 0.f, 0.f};
#pragma unroll
    for (int kk = 0; kk < 128; kk += 32) {
        v16bf a  = a_frag_f32(W2b, 16 * w, 128, kk, lane);       // A[m][k=d]
        v16bf bb = b_frag_bf16(&qt[0][0], 0, 128, kk, lane);     // B[k=d][n=t]
        acc2 = wmma_bf16(a, bb, acc2);
    }
#pragma unroll
    for (int r = 0; r < 8; ++r) {
        const int m = 16 * w + r + hi;
        const int t = t0 + tl;
        out[((size_t)b * 128 + m) * T + t] = tanhf(acc2[r] + ba[m]);
    }
}

// ---------------------------------------------------------------------------
extern "C" void kernel_launch(void* const* d_in, const int* in_sizes, int n_in,
                              void* d_out, int out_size, void* d_ws, size_t ws_size,
                              hipStream_t stream) {
    const float* query = (const float*)d_in[0];
    const float* key_  = (const float*)d_in[1];
    const float* value = (const float*)d_in[2];
    const float* Wq    = (const float*)d_in[3];
    const float* bq    = (const float*)d_in[4];
    const float* Wk    = (const float*)d_in[5];
    const float* bk    = (const float*)d_in[6];
    const float* Wv    = (const float*)d_in[7];
    const float* bv    = (const float*)d_in[8];
    const float* Wa    = (const float*)d_in[9];
    const float* ba    = (const float*)d_in[10];
    float* out = (float*)d_out;

    const int B = 8, T = 4096;
    char* ws = (char*)d_ws;
    const size_t kv_bytes = (size_t)B * 128 * T * sizeof(bf16_t);   // 8 MiB each
    bf16_t* kbf   = (bf16_t*)ws;
    bf16_t* vbf   = (bf16_t*)(ws + kv_bytes);
    float*  Spart = (float*)(ws + 2 * kv_bytes);                    // 8 x B x 128 x 128
    float*  S     = Spart + (size_t)8 * B * 128 * 128;
    float*  W2    = S + (size_t)B * 128 * 128;

    dim3 blk(256);
    dense_tanh_bf16_kernel<128><<<dim3(T / 16, B), blk, 0, stream>>>(Wk, bk, key_,  kbf, T);
    dense_tanh_bf16_kernel<128><<<dim3(T / 16, B), blk, 0, stream>>>(Wv, bv, value, vbf, T);
    s_partial_kernel<<<dim3(8, B, 8), blk, 0, stream>>>(kbf, vbf, Spart, T, T / 8);
    s_reduce_kernel<<<dim3((B * 128 * 128 + 255) / 256), blk, 0, stream>>>(Spart, S, B * 128 * 128);
    w2_kernel<<<dim3(64, B), blk, 0, stream>>>(Wa, S, W2);
    attn_out_kernel<<<dim3(T / 16, B), blk, 0, stream>>>(Wq, bq, query, W2, ba, out, T);
}
